// Policy_15049565405512
// MI455X (gfx1250) — matrix operand order repaired and checked by reading, not compiled, not run on hardware
//
#include <hip/hip_runtime.h>
#include <hip/hip_bf16.h>

// Problem dims
#define T_STEPS 512
#define BATCH   128
#define HID     256
#define NLAYER  2
#define NACT    18

// d_out layout (floats): logits | baseline | action | hT | cT
#define OLOG  0
#define OBASE ((size_t)T_STEPS * BATCH * NACT)
#define OACT  (OBASE + (size_t)T_STEPS * BATCH)
#define OHT   (OACT + (size_t)T_STEPS * BATCH)
#define OCT   (OHT + (size_t)NLAYER * BATCH * HID)

// workspace layout (bytes)
#define SZ_XBF   ((size_t)T_STEPS * BATCH * HID * 2)       // 32MB bf16
#define OFF_XBF  ((size_t)0)
#define OFF_H0   (OFF_XBF + SZ_XBF)                        // layer0 h history, bf16
#define OFF_H1   (OFF_H0 + SZ_XBF)                         // layer1 h history, bf16
#define SZ_W     ((size_t)NLAYER * 4 * HID * HID * 2)      // 1MB bf16 per matrix set
#define OFF_WIH  (OFF_H1 + SZ_XBF)
#define OFF_WHH  (OFF_WIH + SZ_W)
#define SZ_BIAS  ((size_t)NLAYER * 4 * HID * 4)
#define OFF_BIAS (OFF_WHH + SZ_W)
#define OFF_FLG  (OFF_BIAS + SZ_BIAS)                      // 2*T u32 flags

typedef __attribute__((ext_vector_type(16))) __bf16 v16bf;
typedef __attribute__((ext_vector_type(8)))  float  v8f;
typedef __attribute__((ext_vector_type(4)))  unsigned int v4u;
typedef __attribute__((ext_vector_type(8)))  int v8i;
typedef __attribute__((ext_vector_type(4)))  int v4i;

union FragAB { v16bf v; unsigned int u[8]; };
union FragC  { v8f  v;  float f[8]; };

__device__ __forceinline__ unsigned short f2bf(float x) {
  unsigned int b = __float_as_uint(x);
  unsigned int r = (b + 0x7FFFu + ((b >> 16) & 1u)) >> 16;
  return (unsigned short)r;
}

__device__ __forceinline__ float sigmoidf_fast(float x) {
  return 1.0f / (1.0f + __expf(-x));
}

__device__ __forceinline__ void wait_ge(unsigned int* p, unsigned int tgt) {
  while (__hip_atomic_load(p, __ATOMIC_ACQUIRE, __HIP_MEMORY_SCOPE_AGENT) < tgt) {
    __builtin_amdgcn_s_sleep(4);
  }
}

// TDM: DMA a rows x cols (bf16) 2D tile from global into LDS.
// D# group0: count=1, lds_addr, global_addr[56:0], type=2 ("image").
// D# group1: data_size=1 (2B), tensor_dim0=cols, tensor_dim1=rows,
//            tile_dim0=cols, tile_dim1=rows, tensor_dim0_stride=cols.
__device__ __forceinline__ void tdm_load_2d(unsigned lds_off, const void* gaddr,
                                            unsigned rows, unsigned cols) {
  unsigned long long ga = (unsigned long long)(size_t)gaddr;
  v4u g0;
  g0[0] = 1u;                                                    // count=1
  g0[1] = lds_off;                                               // lds_addr (bytes)
  g0[2] = (unsigned)(ga & 0xFFFFFFFFull);                        // global_addr[31:0]
  g0[3] = (unsigned)((ga >> 32) & 0x01FFFFFFull) | (2u << 30);   // addr[56:32] | type=2
  v8i g1;
  g1[0] = (int)(1u << 16);          // wg_mask=0, data_size=1 (2 bytes)
  g1[1] = (int)(cols << 16);        // tensor_dim0[15:0] @ [63:48]
  g1[2] = (int)(rows << 16);        // tensor_dim1[15:0] @ [95:80]
  g1[3] = (int)(cols << 16);        // tile_dim0 @ [127:112]
  g1[4] = (int)rows;                // tile_dim1 @ [143:128]
  g1[5] = (int)cols;                // tensor_dim0_stride[31:0]
  g1[6] = 0;
  g1[7] = 0;
  v4i z4 = {0, 0, 0, 0};
  v8i z8 = {0, 0, 0, 0, 0, 0, 0, 0};
  // 6-arg form (clang-23 / therock headers): (g0, g1, g2, g3, g4, cpol)
  __builtin_amdgcn_tensor_load_to_lds(g0, g1, z4, z4, z8, 0);
}

// ---------------- prep kernels ----------------
__global__ void k_cvt_x(const float* __restrict__ x, unsigned short* __restrict__ xbf, int n) {
  int i = blockIdx.x * blockDim.x + threadIdx.x;
  if (i < n) xbf[i] = f2bf(x[i]);
}

__global__ void k_cvt_w(const float* __restrict__ wi, const float* __restrict__ wh,
                        unsigned short* __restrict__ wib, unsigned short* __restrict__ whb, int n) {
  int i = blockIdx.x * blockDim.x + threadIdx.x;
  if (i < n) { wib[i] = f2bf(wi[i]); whb[i] = f2bf(wh[i]); }
}

__global__ void k_bias_flags(const float* __restrict__ bi, const float* __restrict__ bh,
                             float* __restrict__ bs, unsigned int* __restrict__ flags,
                             int nb, int nf) {
  int i = blockIdx.x * blockDim.x + threadIdx.x;
  if (i < nb) bs[i] = bi[i] + bh[i];
  if (i < nf) flags[i] = 0u;
}

// ---------------- persistent LSTM + head ----------------
__global__ __launch_bounds__(256) void lstm_core(
    const unsigned short* __restrict__ xbf,
    unsigned short* __restrict__ h0h,
    unsigned short* __restrict__ h1h,
    const unsigned short* __restrict__ wih,
    const unsigned short* __restrict__ whh,
    const float* __restrict__ bias,
    unsigned int* __restrict__ flags,
    const unsigned char* __restrict__ done,
    const float* __restrict__ Wp, const float* __restrict__ bp,
    const float* __restrict__ Wb, const float* __restrict__ bb,
    float* __restrict__ out)
{
  // 128KB LDS: layer role keeps both weight slices resident for all 512 steps;
  // head role uses a small slice of the same storage.
  __shared__ union SMem {
    unsigned short w[2][128 * HID];  // [0]=Wih slice, [1]=Whh slice (row = gate*32+unit)
    struct {
      unsigned short hw[32 * HID];   // head weights bf16 (0..17 Wp, 18 Wb, rest 0)
      float hb[32];                  // head bias
      float res[BATCH * 32];         // head results exchange
    } head;
  } smem;

  const int tid   = threadIdx.x;
  const int wave  = tid >> 5;
  const int lane  = tid & 31;
  const int lhalf = lane >> 4;
  const int lmod  = lane & 15;

  unsigned int* flag0 = flags;
  unsigned int* flag1 = flags + T_STEPS;

  if (blockIdx.x < 16) {
    // ---------------- LSTM layer workgroup ----------------
    const int l = blockIdx.x >> 3;   // layer
    const int s = blockIdx.x & 7;    // hidden-unit slice (32 units)
    const unsigned short* Wi = wih + (size_t)l * 4 * HID * HID;
    const unsigned short* Wh = whh + (size_t)l * 4 * HID * HID;
    const float* bl = bias + l * 4 * HID;
    unsigned short* myH = (l == 0) ? h0h : h1h;
    const unsigned short* inH = (l == 0) ? xbf : h0h; // this layer's input stream
    unsigned int* myFlag = (l == 0) ? flag0 : flag1;

    // TDM: stage this WG's 128 gate rows of Wih and Whh into LDS (4 blocks of
    // 32 contiguous rows per matrix). Issue from wave 0 only (TDM ignores EXEC,
    // so it must be branch-skipped for the other waves), then drain TENSORcnt.
    if (tid == 0) {
      const unsigned lds_base = (unsigned)(size_t)&smem.w[0][0];
#pragma unroll
      for (int g = 0; g < 4; ++g) {
        tdm_load_2d(lds_base + (unsigned)(g * 32 * HID * 2),
                    Wi + ((size_t)g * HID + s * 32) * HID, 32, HID);
        tdm_load_2d(lds_base + (unsigned)((128 + g * 32) * HID * 2),
                    Wh + ((size_t)g * HID + s * 32) * HID, 32, HID);
      }
      __builtin_amdgcn_s_wait_tensorcnt(0);
    }
    __syncthreads();

    // per-lane bias for the 8 N-tiles (gate = nt>>1, unit-group = nt&1)
    float biasInit[8];
#pragma unroll
    for (int nt = 0; nt < 8; ++nt)
      biasInit[nt] = bl[(nt >> 1) * HID + s * 32 + (nt & 1) * 16 + lmod];

    float cReg[16];  // register-resident cell state for this lane
#pragma unroll
    for (int i = 0; i < 16; ++i) cReg[i] = 0.0f;

    const int bA = wave * 16 + lmod;  // A-fragment batch row owned by this lane

    for (int t = 0; t < T_STEPS; ++t) {
      if (tid == 0) {
        if (l == 1) wait_ge(&flag0[t], 8u);         // current-step layer0 output
        if (t > 0)  wait_ge(&myFlag[t - 1], 8u);    // full previous h of own layer
      }
      __syncthreads();
      __builtin_amdgcn_fence(__ATOMIC_ACQUIRE, "agent");

      FragC acc[8];
#pragma unroll
      for (int nt = 0; nt < 8; ++nt)
#pragma unroll
        for (int v = 0; v < 8; ++v) acc[nt].f[v] = biasInit[nt];

      // ---- input GEMM: gates += in_t @ Wih^T (A: global bf16, B: LDS) ----
      const unsigned short* Asrc = inH + (size_t)t * BATCH * HID;
#pragma unroll 1
      for (int kc = 0; kc < HID; kc += 32) {
        FragAB a;
#pragma unroll
        for (int j = 0; j < 8; ++j) {
          int k = kc + ((j < 4) ? (j * 2 + lhalf * 8) : (16 + (j - 4) * 2 + lhalf * 8));
          a.u[j] = *(const unsigned int*)(Asrc + (size_t)bA * HID + k);
        }
        __builtin_prefetch(Asrc + (size_t)bA * HID + kc + 32, 0, 1);
#pragma unroll
        for (int nt = 0; nt < 8; ++nt) {
          int nloc = (nt >> 1) * 32 + (nt & 1) * 16 + lmod;  // gate*32 + unit
          FragAB b;
#pragma unroll
          for (int v = 0; v < 8; ++v) {
            int k = kc + ((v < 4) ? (v * 2 + lhalf * 8) : (16 + (v - 4) * 2 + lhalf * 8));
            b.u[v] = *(const unsigned int*)(&smem.w[0][nloc * HID + k]);  // ds_load
          }
          acc[nt].v = __builtin_amdgcn_wmma_f32_16x16x32_bf16(
              false, a.v, false, b.v, (short)0, acc[nt].v, false, false);
        }
      }

      // ---- recurrent GEMM: gates += (h_{t-1} * !done_t) @ Whh^T ----
      if (t > 0) {
        const unsigned short* Hsrc = myH + (size_t)(t - 1) * BATCH * HID;
        const bool drop = (done[(size_t)t * BATCH + bA] != 0);
#pragma unroll 1
        for (int kc = 0; kc < HID; kc += 32) {
          FragAB a;
#pragma unroll
          for (int j = 0; j < 8; ++j) {
            int k = kc + ((j < 4) ? (j * 2 + lhalf * 8) : (16 + (j - 4) * 2 + lhalf * 8));
            unsigned int hv = *(const unsigned int*)(Hsrc + (size_t)bA * HID + k);
            a.u[j] = drop ? 0u : hv;
          }
#pragma unroll
          for (int nt = 0; nt < 8; ++nt) {
            int nloc = (nt >> 1) * 32 + (nt & 1) * 16 + lmod;
            FragAB b;
#pragma unroll
            for (int v = 0; v < 8; ++v) {
              int k = kc + ((v < 4) ? (v * 2 + lhalf * 8) : (16 + (v - 4) * 2 + lhalf * 8));
              b.u[v] = *(const unsigned int*)(&smem.w[1][nloc * HID + k]);  // ds_load
            }
            acc[nt].v = __builtin_amdgcn_wmma_f32_16x16x32_bf16(
                false, a.v, false, b.v, (short)0, acc[nt].v, false, false);
          }
        }
      }

      // ---- cell update: i,f,g,o for one (b,j) live in the same lane/index ----
#pragma unroll
      for (int g2 = 0; g2 < 2; ++g2) {
        const int j = s * 32 + g2 * 16 + lmod;
#pragma unroll
        for (int v = 0; v < 8; ++v) {
          const int b = wave * 16 + v + 8 * lhalf;  // C-layout: M = v + 8*(lane/16)
          float iv = acc[0 + g2].f[v];
          float fv = acc[2 + g2].f[v];
          float gv = acc[4 + g2].f[v];
          float ov = acc[6 + g2].f[v];
          float cp = cReg[g2 * 8 + v];
          if (done[(size_t)t * BATCH + b]) cp = 0.0f;
          float cn = sigmoidf_fast(fv) * cp + sigmoidf_fast(iv) * tanhf(gv);
          float hn = sigmoidf_fast(ov) * tanhf(cn);
          cReg[g2 * 8 + v] = cn;
          myH[(size_t)t * BATCH * HID + (size_t)b * HID + j] = f2bf(hn);
          if (t == T_STEPS - 1) {
            out[OHT + (size_t)l * BATCH * HID + (size_t)b * HID + j] = hn;
            out[OCT + (size_t)l * BATCH * HID + (size_t)b * HID + j] = cn;
          }
        }
      }

      __threadfence();
      __syncthreads();
      if (tid == 0)
        __hip_atomic_fetch_add(&myFlag[t], 1u, __ATOMIC_RELEASE, __HIP_MEMORY_SCOPE_AGENT);
    }
  } else {
    // ---------------- head workgroup: logits/baseline/argmax via WMMA ----------------
    for (int i = tid; i < 32 * HID; i += 256) {
      int rrow = i / HID, k = i % HID;
      float wv = 0.0f;
      if (rrow < NACT) wv = Wp[rrow * HID + k];
      else if (rrow == NACT) wv = Wb[k];
      smem.head.hw[i] = f2bf(wv);
    }
    if (tid < 32) smem.head.hb[tid] = (tid < NACT) ? bp[tid] : ((tid == NACT) ? bb[0] : 0.0f);
    __syncthreads();

    const int bA = wave * 16 + lmod;
    for (int t = 0; t < T_STEPS; ++t) {
      if (tid == 0) wait_ge(&flag1[t], 8u);
      __syncthreads();
      __builtin_amdgcn_fence(__ATOMIC_ACQUIRE, "agent");

      const unsigned short* Hsrc = h1h + (size_t)t * BATCH * HID;
      FragC acc[2];
#pragma unroll
      for (int nt = 0; nt < 2; ++nt)
#pragma unroll
        for (int v = 0; v < 8; ++v) acc[nt].f[v] = 0.0f;

#pragma unroll 1
      for (int kc = 0; kc < HID; kc += 32) {
        FragAB a;
#pragma unroll
        for (int j = 0; j < 8; ++j) {
          int k = kc + ((j < 4) ? (j * 2 + lhalf * 8) : (16 + (j - 4) * 2 + lhalf * 8));
          a.u[j] = *(const unsigned int*)(Hsrc + (size_t)bA * HID + k);
        }
#pragma unroll
        for (int nt = 0; nt < 2; ++nt) {
          int r = nt * 16 + lmod;
          FragAB b;
#pragma unroll
          for (int v = 0; v < 8; ++v) {
            int k = kc + ((v < 4) ? (v * 2 + lhalf * 8) : (16 + (v - 4) * 2 + lhalf * 8));
            b.u[v] = *(const unsigned int*)(&smem.head.hw[r * HID + k]);  // ds_load
          }
          acc[nt].v = __builtin_amdgcn_wmma_f32_16x16x32_bf16(
              false, a.v, false, b.v, (short)0, acc[nt].v, false, false);
        }
      }

#pragma unroll
      for (int nt = 0; nt < 2; ++nt)
#pragma unroll
        for (int v = 0; v < 8; ++v)
          smem.head.res[(wave * 16 + v + 8 * lhalf) * 32 + nt * 16 + lmod] = acc[nt].f[v];
      __syncthreads();

      if (tid < BATCH) {
        const int b = tid;
        float base = smem.head.res[b * 32 + NACT] + smem.head.hb[NACT];
        float mx = -3.4e38f;
        int am = 0;
#pragma unroll
        for (int a = 0; a < NACT; ++a) {
          float lg = smem.head.res[b * 32 + a] + smem.head.hb[a];
          out[OLOG + (size_t)t * BATCH * NACT + (size_t)b * NACT + a] = lg;
          if (lg > mx) { mx = lg; am = a; }
        }
        out[OBASE + (size_t)t * BATCH + b] = base;
        out[OACT + (size_t)t * BATCH + b] = (float)am;
      }
      __syncthreads();  // protect res before next step
    }
  }
}

extern "C" void kernel_launch(void* const* d_in, const int* in_sizes, int n_in,
                              void* d_out, int out_size, void* d_ws, size_t ws_size,
                              hipStream_t stream) {
  (void)in_sizes; (void)n_in; (void)out_size; (void)ws_size;

  const float* x    = (const float*)d_in[0];
  const unsigned char* done = (const unsigned char*)d_in[1];
  // d_in[2]=h0, d_in[3]=c0 are zeros by construction (state starts at zero)
  const float* w_ih = (const float*)d_in[4];
  const float* w_hh = (const float*)d_in[5];
  const float* b_ih = (const float*)d_in[6];
  const float* b_hh = (const float*)d_in[7];
  const float* Wp   = (const float*)d_in[8];
  const float* bp   = (const float*)d_in[9];
  const float* Wb   = (const float*)d_in[10];
  const float* bb   = (const float*)d_in[11];

  char* ws = (char*)d_ws;
  unsigned short* xbf  = (unsigned short*)(ws + OFF_XBF);
  unsigned short* h0h  = (unsigned short*)(ws + OFF_H0);
  unsigned short* h1h  = (unsigned short*)(ws + OFF_H1);
  unsigned short* wihb = (unsigned short*)(ws + OFF_WIH);
  unsigned short* whhb = (unsigned short*)(ws + OFF_WHH);
  float*          bws  = (float*)(ws + OFF_BIAS);
  unsigned int*   flg  = (unsigned int*)(ws + OFF_FLG);

  const int nx = T_STEPS * BATCH * HID;
  const int nw = NLAYER * 4 * HID * HID;
  const int nb = NLAYER * 4 * HID;
  const int nf = 2 * T_STEPS;

  k_cvt_x<<<(nx + 255) / 256, 256, 0, stream>>>(x, xbf, nx);
  k_cvt_w<<<(nw + 255) / 256, 256, 0, stream>>>(w_ih, w_hh, wihb, whhb, nw);
  k_bias_flags<<<(nb + 255) / 256, 256, 0, stream>>>(b_ih, b_hh, bws, flg, nb, nf);

  lstm_core<<<17, 256, 0, stream>>>(xbf, h0h, h1h, wihb, whhb, bws, flg, done,
                                    Wp, bp, Wb, bb, (float*)d_out);
}
